// NgramMultiheadAttention_9113920602674
// MI455X (gfx1250) — compile-verified, matrix operand order checked by reading, not run on hardware
//
#include <hip/hip_runtime.h>

// ---------------- problem constants ----------------
#define TT     512          // REAL_T
#define NB_T   4            // BSZ
#define NH     16           // heads
#define HD     64           // head dim
#define EDIM   1024         // embed dim
#define QKVLD  3072         // 3*E
#define ROWS   6144         // tgt_len * BSZ = 1536*4
#define NBH    512          // NUM_BUCKETS * NH

typedef __attribute__((ext_vector_type(16))) _Float16 v16h;
typedef __attribute__((ext_vector_type(8)))  _Float16 v8h;
typedef __attribute__((ext_vector_type(8)))  float    v8f;

static __device__ inline v16h wmma_pack(v8h lo, v8h hi) {
    v16h a;
#pragma unroll
    for (int i = 0; i < 8; ++i) { a[i] = lo[i]; a[8 + i] = hi[i]; }
    return a;
}

// 16x32 A-tile / 32x16 B-tile loader from a row-major f16 matrix.
// Operand row (M for A, N for B) = row0 + lane%16; K contiguous from k0.
// ISA lane layout: lanes 0-15 halves -> K {0..7,16..23}; lanes 16-31 -> K {8..15,24..31}.
static __device__ inline v16h load_tile(const _Float16* __restrict__ p, long ld,
                                        int row0, int k0) {
    const int lane = threadIdx.x & 31;
    const _Float16* src = p + (long)(row0 + (lane & 15)) * ld + k0 + ((lane < 16) ? 0 : 8);
    return wmma_pack(*(const v8h*)src, *(const v8h*)(src + 16));
}

static __device__ inline v8f wmma_f16(v16h a, v16h b, v8f c) {
    return __builtin_amdgcn_wmma_f32_16x16x32_f16(false, a, false, b, (short)0, c,
                                                  false, false);
}

// ---------------- fp32 -> f16 convert ----------------
__global__ void cvt_f32_f16(const float* __restrict__ in, _Float16* __restrict__ out,
                            long n) {
    long i = (long)blockIdx.x * blockDim.x + threadIdx.x;
    long stride = (long)gridDim.x * blockDim.x;
    for (; i < n; i += stride) out[i] = (_Float16)in[i];
}

// ---------------- generic WMMA GEMM: C = A[MxK] * B[NxK]^T + bias ----------------
// Each wave owns a 32x32 macro-tile (2x2 WMMA accumulators): per K-step it loads
// 2 A-tiles + 2 B-tiles (4 KB) and issues 4 WMMAs (64 KFLOP) -> 16 FLOP/byte of
// L2 traffic (2x the naive wave-per-tile scheme).  8 consecutive waves share the
// same macro-tile row so A panels stay hot in WGP$/L2 (192 MB easily holds them).
// out_f16: store f16 else f32.  Columns n < scale_nlim are scaled by `scale`
// after bias add (used for q * hd^-0.5 in the qkv projection).
__global__ __launch_bounds__(256) void gemm_f16_wmma(
    const _Float16* __restrict__ A, const _Float16* __restrict__ B,
    const float* __restrict__ bias, void* __restrict__ Cout,
    int M, int N, int K, int out_f16, float scale, int scale_nlim)
{
    const int wave = threadIdx.x >> 5, lane = threadIdx.x & 31;
    const int ntN = N >> 5;                      // 32-wide macro-tiles along N
    long tile = (long)blockIdx.x * 8 + wave;
    long mt = tile / ntN;
    int  nt = (int)(tile % ntN);
    if (mt >= (long)(M >> 5)) return;            // wave-uniform
    const int m0 = (int)mt << 5, n0 = nt << 5;

    v8f acc00 = {}, acc01 = {}, acc10 = {}, acc11 = {};
    for (int k0 = 0; k0 < K; k0 += 32) {
        if (k0 + 32 < K) {                        // GLOBAL_PREFETCH of next K chunk
            __builtin_prefetch(A + (long)(m0 + (lane & 15)) * K + k0 + 32, 0, 3);
            __builtin_prefetch(A + (long)(m0 + 16 + (lane & 15)) * K + k0 + 32, 0, 3);
            __builtin_prefetch(B + (long)(n0 + (lane & 15)) * K + k0 + 32, 0, 3);
            __builtin_prefetch(B + (long)(n0 + 16 + (lane & 15)) * K + k0 + 32, 0, 3);
        }
        v16h a0 = load_tile(A, K, m0,      k0);
        v16h a1 = load_tile(A, K, m0 + 16, k0);
        v16h b0 = load_tile(B, K, n0,      k0);
        v16h b1 = load_tile(B, K, n0 + 16, k0);
        acc00 = wmma_f16(a0, b0, acc00);
        acc01 = wmma_f16(a0, b1, acc01);
        acc10 = wmma_f16(a1, b0, acc10);
        acc11 = wmma_f16(a1, b1, acc11);
    }

    const int lcol  = lane & 15;
    const int mbase = (lane < 16) ? 0 : 8;
#pragma unroll
    for (int im = 0; im < 2; ++im) {
#pragma unroll
        for (int in = 0; in < 2; ++in) {
            v8f acc = (im == 0) ? ((in == 0) ? acc00 : acc01)
                                : ((in == 0) ? acc10 : acc11);
            const int n  = n0 + in * 16 + lcol;
            const int mb = m0 + im * 16 + mbase;
            const float bv = bias ? bias[n] : 0.0f;
            const float sc = (n < scale_nlim) ? scale : 1.0f;
#pragma unroll
            for (int r = 0; r < 8; ++r) {
                float v = (acc[r] + bv) * sc;
                if (out_f16) ((_Float16*)Cout)[(long)(mb + r) * N + n] = (_Float16)v;
                else         ((float*)Cout)[(long)(mb + r) * N + n] = v;
            }
        }
    }
}

// ---------------- fused attention: scores + rel gather + softmax + PV ----------------
// grid: (TT/16, 64, 3). mode 0 = main stream (S=512), mode 1..2 = ngram (S=1024).
// One workgroup handles 16 query rows of one (b,h). Scores stay register-resident
// (each wave owns 16x(S/8)); probabilities go to LDS as f16 for the PV WMMA.
// This keeps the 64/256 MB score matrices out of HBM entirely (the naive
// formulation would move ~0.7 GB; fused, the attention phase streams ~60 MB).
__global__ __launch_bounds__(256) void attn_softmax_kernel(
    const _Float16* __restrict__ qkv,   // ROWS x 3072 (q scaled)
    const float*    __restrict__ vals,  // ROWS x 512  (bucket values, fp32)
    const int*      __restrict__ ib_main,  // 4 x 512 x 512
    const int*      __restrict__ ib_rel,   // 4 x 512 x 1024
    const float*    __restrict__ mask_main,// 512 x 512
    const float*    __restrict__ mask_ng,  // 2 x 512 x 1024
    _Float16*       __restrict__ attn)     // ROWS x 1024
{
    const int mode = blockIdx.z;
    const int bh = blockIdx.y;
    const int b = bh >> 4, h = bh & 15;
    const int t0 = blockIdx.x << 4;
    const int S   = (mode == 0) ? TT : 2 * TT;
    const int TPW = S >> 7;                       // 16-col tiles per wave (4 or 8)
    const int wave = threadIdx.x >> 5, lane = threadIdx.x & 31;
    const int qrow0 = (mode == 0) ? t0 : (TT + (mode - 1) * TT + t0);
    const int mbase = (lane < 16) ? 0 : 8;
    const int lrow  = lane & 15;
    const int klo   = (lane < 16) ? 0 : 8;

    __shared__ float    red[8][16];
    __shared__ _Float16 P[16][2 * TT + 8];        // padded to dodge bank conflicts

    // ---- preload 16x64 Q block as two 16x32 A tiles ----
    v16h aq[2];
    {
        const _Float16* qb = qkv + ((long)(qrow0 + lrow) * NB_T + b) * QKVLD + h * HD;
#pragma unroll
        for (int kk = 0; kk < 2; ++kk) {
            const _Float16* src = qb + kk * 32 + klo;
            aq[kk] = wmma_pack(*(const v8h*)src, *(const v8h*)(src + 16));
        }
    }

    // ---- phase 1: scores (register resident) + rel bias gather + mask ----
    v8f acc[8];
    for (int j = 0; j < TPW; ++j) {
        const int s0   = (wave + j * 8) << 4;
        const int skey = s0 + lrow;
        const int srow = (mode == 0 || skey < TT) ? skey
                       : (TT + (mode - 1) * TT + (skey - TT));
        const _Float16* kb = qkv + ((long)srow * NB_T + b) * QKVLD + EDIM + h * HD;
        v8f c = {};
#pragma unroll
        for (int kk = 0; kk < 2; ++kk) {
            const _Float16* src = kb + kk * 32 + klo;
            v16h bt = wmma_pack(*(const v8h*)src, *(const v8h*)(src + 16));
            c = wmma_f16(aq[kk], bt, c);
        }
#pragma unroll
        for (int r = 0; r < 8; ++r) {
            const int t = t0 + mbase + r;
            const int n = s0 + lrow;               // this lane's output column
            float add;
            if (mode == 0) {
                int bucket = ib_main[((long)b * TT + t) * TT + n];
                add = vals[((long)t * NB_T + b) * NBH + bucket * NH + h]
                    + mask_main[t * TT + n];
            } else {
                int qr = TT + (mode - 1) * TT + t;
                int bucket = ib_rel[((long)b * TT + t) * (2 * TT) + n];
                add = vals[((long)qr * NB_T + b) * NBH + bucket * NH + h]
                    + mask_ng[((long)(mode - 1) * TT + t) * (2 * TT) + n];
            }
            c[r] += add;
        }
        acc[j] = c;
    }

    // ---- phase 2: row max (lanes within 16-lane half, then across waves) ----
    float fmx[8], fsum[8];
#pragma unroll
    for (int r = 0; r < 8; ++r) {
        float m = -3.0e38f;
        for (int j = 0; j < TPW; ++j) m = fmaxf(m, acc[j][r]);
#pragma unroll
        for (int off = 1; off <= 8; off <<= 1) m = fmaxf(m, __shfl_xor(m, off, 32));
        fmx[r] = m;
    }
    if (lane == 0 || lane == 16) {
#pragma unroll
        for (int r = 0; r < 8; ++r) red[wave][mbase + r] = fmx[r];
    }
    __syncthreads();
#pragma unroll
    for (int r = 0; r < 8; ++r) {
        float m = red[0][mbase + r];
#pragma unroll
        for (int w = 1; w < 8; ++w) m = fmaxf(m, red[w][mbase + r]);
        fmx[r] = m;
    }
    __syncthreads();

    // ---- phase 3: exp + row sum ----
#pragma unroll
    for (int r = 0; r < 8; ++r) {
        float s = 0.0f;
        for (int j = 0; j < TPW; ++j) {
            float e = __expf(acc[j][r] - fmx[r]);
            acc[j][r] = e;
            s += e;
        }
#pragma unroll
        for (int off = 1; off <= 8; off <<= 1) s += __shfl_xor(s, off, 32);
        fsum[r] = s;
    }
    if (lane == 0 || lane == 16) {
#pragma unroll
        for (int r = 0; r < 8; ++r) red[wave][mbase + r] = fsum[r];
    }
    __syncthreads();
#pragma unroll
    for (int r = 0; r < 8; ++r) {
        float s = 0.0f;
#pragma unroll
        for (int w = 0; w < 8; ++w) s += red[w][mbase + r];
        fsum[r] = 1.0f / s;
    }

    // ---- phase 4: normalized probs -> LDS f16 ----
    for (int j = 0; j < TPW; ++j) {
        const int n = ((wave + j * 8) << 4) + lrow;
#pragma unroll
        for (int r = 0; r < 8; ++r)
            P[mbase + r][n] = (_Float16)(acc[j][r] * fsum[r]);
    }
    __syncthreads();

    // ---- phase 5: attn = P @ V  (waves 0..3 each own one 16-col d tile) ----
    if (wave < 4) {
        const int d = (wave << 4) + lrow;
        v8f o = {};
        for (int k0 = 0; k0 < S; k0 += 32) {
            v16h a, bt;
#pragma unroll
            for (int i = 0; i < 8; ++i) {
                a[i]     = P[lrow][k0 + klo + i];
                a[8 + i] = P[lrow][k0 + klo + 16 + i];
                int s1 = k0 + klo + i;
                int s2 = k0 + klo + 16 + i;
                int sr1 = (mode == 0 || s1 < TT) ? s1 : (TT + (mode - 1) * TT + (s1 - TT));
                int sr2 = (mode == 0 || s2 < TT) ? s2 : (TT + (mode - 1) * TT + (s2 - TT));
                bt[i]     = qkv[((long)sr1 * NB_T + b) * QKVLD + 2 * EDIM + h * HD + d];
                bt[8 + i] = qkv[((long)sr2 * NB_T + b) * QKVLD + 2 * EDIM + h * HD + d];
            }
            o = wmma_f16(a, bt, o);
        }
#pragma unroll
        for (int r = 0; r < 8; ++r) {
            const int grow = qrow0 + mbase + r;
            attn[((long)grow * NB_T + b) * EDIM + h * HD + d] = (_Float16)o[r];
        }
    }
}

// ---------------- host-side orchestration ----------------
extern "C" void kernel_launch(void* const* d_in, const int* in_sizes, int n_in,
                              void* d_out, int out_size, void* d_ws, size_t ws_size,
                              hipStream_t stream) {
    (void)in_sizes; (void)n_in; (void)out_size; (void)ws_size;
    const float* query     = (const float*)d_in[0];
    const float* mask_main = (const float*)d_in[1];
    const float* mask_ng   = (const float*)d_in[2];
    const int*   ib_main   = (const int*)d_in[3];
    const int*   ib_rel    = (const int*)d_in[4];
    const float* Win       = (const float*)d_in[5];
    const float* bin       = (const float*)d_in[6];
    const float* Wout      = (const float*)d_in[7];
    const float* bout      = (const float*)d_in[8];
    const float* RW        = (const float*)d_in[9];
    const float* rb        = (const float*)d_in[10];

    auto a256 = [](size_t x) { return (x + 255) & ~(size_t)255; };
    char* p = (char*)d_ws;
    _Float16* Xh    = (_Float16*)p; p += a256((size_t)ROWS * EDIM * 2);
    _Float16* Winh  = (_Float16*)p; p += a256((size_t)3 * EDIM * EDIM * 2);
    _Float16* RWh   = (_Float16*)p; p += a256((size_t)NBH * EDIM * 2);
    _Float16* Wouth = (_Float16*)p; p += a256((size_t)EDIM * EDIM * 2);
    _Float16* qkvh  = (_Float16*)p; p += a256((size_t)ROWS * QKVLD * 2);
    float*    valsf = (float*)p;    p += a256((size_t)ROWS * NBH * 4);
    _Float16* attnh = (_Float16*)p; p += a256((size_t)ROWS * EDIM * 2);

    cvt_f32_f16<<<4096, 256, 0, stream>>>(query, Xh,    (long)ROWS * EDIM);
    cvt_f32_f16<<<4096, 256, 0, stream>>>(Win,   Winh,  (long)3 * EDIM * EDIM);
    cvt_f32_f16<<<2048, 256, 0, stream>>>(RW,    RWh,   (long)NBH * EDIM);
    cvt_f32_f16<<<2048, 256, 0, stream>>>(Wout,  Wouth, (long)EDIM * EDIM);

    // qkv = X @ Win^T + bin, q-part (cols < E) scaled by hd^-0.5 = 0.125, f16 out
    {
        long tiles = (long)(ROWS / 32) * (QKVLD / 32);
        gemm_f16_wmma<<<dim3((unsigned)((tiles + 7) / 8)), 256, 0, stream>>>(
            Xh, Winh, bin, (void*)qkvh, ROWS, QKVLD, EDIM, 1, 0.125f, EDIM);
    }
    // vals = X @ RW^T + rb, fp32 out (feeds the bucket gather)
    {
        long tiles = (long)(ROWS / 32) * (NBH / 32);
        gemm_f16_wmma<<<dim3((unsigned)((tiles + 7) / 8)), 256, 0, stream>>>(
            Xh, RWh, rb, (void*)valsf, ROWS, NBH, EDIM, 0, 1.0f, 0);
    }
    // fused attention: main stream + 2 ngram streams
    attn_softmax_kernel<<<dim3(TT / 16, 64, 3), 256, 0, stream>>>(
        qkvh, valsf, ib_main, ib_rel, mask_main, mask_ng, attnh);
    // out = attn @ Wout^T + bout, fp32 out
    {
        long tiles = (long)(ROWS / 32) * (EDIM / 32);
        gemm_f16_wmma<<<dim3((unsigned)((tiles + 7) / 8)), 256, 0, stream>>>(
            attnh, Wouth, bout, d_out, ROWS, EDIM, EDIM, 0, 1.0f, 0);
    }
}
